// SpatialColorAE_81329500717511
// MI455X (gfx1250) — matrix operand order repaired and checked by reading, not compiled, not run on hardware
//
#include <hip/hip_runtime.h>
#include <hip/hip_bf16.h>

#define NODES 65536
#define EDGES 524288
#define HIDD  128

typedef __bf16 bf16;
typedef __attribute__((ext_vector_type(16))) __bf16 v16bf;
typedef __attribute__((ext_vector_type(8)))  float  v8f;

__device__ __forceinline__ float siluf(float v) { return v * (1.0f / (1.0f + __expf(-v))); }
// order-preserving float<->uint encoding for atomicMax-based segment_max
__device__ __forceinline__ unsigned fenc(float f) {
    unsigned u = __float_as_uint(f);
    return (u & 0x80000000u) ? ~u : (u | 0x80000000u);
}
__device__ __forceinline__ float fdec(unsigned u) {
    return __uint_as_float((u & 0x80000000u) ? (u ^ 0x80000000u) : ~u);
}

// ---------------------------------------------------------------------------
// e = silu(edge_attr @ Wp + bp) stored bf16 (WMMA A-operand for the fused
// edge kernel). K=3 -> plain VALU, bandwidth bound anyway.
// ---------------------------------------------------------------------------
__global__ __launch_bounds__(256) void edge_proj_kernel(
    const float* __restrict__ ea, const float* __restrict__ w,
    const float* __restrict__ b, bf16* __restrict__ e_bf)
{
    long i = (long)blockIdx.x * 256 + threadIdx.x;  // E*128 threads
    long e = i >> 7; int f = (int)(i & 127);
    float v = b[f];
    #pragma unroll
    for (int j = 0; j < 3; ++j) v += ea[e * 3 + j] * w[j * HIDD + f];
    e_bf[i] = (bf16)siluf(v);
}

// ---------------------------------------------------------------------------
// Layer-0 node projections: xl = x@Wl+bl, xr = x@Wr+br  (K=12, VALU)
// ---------------------------------------------------------------------------
__global__ __launch_bounds__(256) void node_proj12_kernel(
    const float* __restrict__ x,
    const float* __restrict__ Wl, const float* __restrict__ bl,
    const float* __restrict__ Wr, const float* __restrict__ br,
    float* __restrict__ xl, float* __restrict__ xr)
{
    long i = (long)blockIdx.x * 256 + threadIdx.x;  // N*128 threads
    long n = i >> 7; int f = (int)(i & 127);
    float vl = bl[f], vr = br[f];
    #pragma unroll
    for (int j = 0; j < 12; ++j) {
        float xv = x[n * 12 + j];
        vl += xv * Wl[j * HIDD + f];
        vr += xv * Wr[j * HIDD + f];
    }
    xl[i] = vl; xr[i] = vr;
}

// ---------------------------------------------------------------------------
// WMMA GEMM: C[M x 128] = act(A_bf16[M x 128] @ W_f32[128 x 128] + bias)
// W transposed+converted to bf16 in LDS once per block. Block = 8 waves,
// wave w owns n-tile w of one 16-row m-tile. 4 k-steps of 16x16x32 bf16 WMMA.
// ACT: 0 = none -> f32 out, 2 = silu -> bf16 out.
// ---------------------------------------------------------------------------
template <int ACT>
__global__ __launch_bounds__(256) void gemm128_wmma_kernel(
    const bf16* __restrict__ A, const float* __restrict__ W,
    const float* __restrict__ bias, float* __restrict__ Cf, bf16* __restrict__ Cb)
{
    __shared__ bf16 WT[HIDD * HIDD];   // [n*128 + k], 32 KB
    __shared__ float biasS[HIDD];
    int tid = threadIdx.x;
    for (int i = tid; i < HIDD * HIDD; i += 256) {
        int k = i >> 7, n = i & 127;
        WT[n * HIDD + k] = (bf16)W[i];
    }
    if (tid < HIDD) biasS[tid] = bias[tid];
    __syncthreads();

    int wave = tid >> 5, lane = tid & 31;
    int ln = lane & 15, half = lane >> 4;
    long m0 = (long)blockIdx.x * 16;
    int n0 = wave * 16;

    const bf16* arow = A + (m0 + ln) * HIDD;
    v8f acc = {0.f, 0.f, 0.f, 0.f, 0.f, 0.f, 0.f, 0.f};
    #pragma unroll
    for (int k = 0; k < 4; ++k) {
        v16bf a, bfr;
        int kb = k * 32 + half * 16;  // A layout: half 0 -> K 0-15, half 1 -> K 16-31
        #pragma unroll
        for (int i = 0; i < 8; ++i) { a[i] = arow[kb + i]; a[i + 8] = arow[kb + 8 + i]; }
        const bf16* bcol = &WT[(n0 + ln) * HIDD + k * 32 + half * 16];
        #pragma unroll
        for (int i = 0; i < 16; ++i) bfr[i] = bcol[i];
        acc = __builtin_amdgcn_wmma_f32_16x16x32_bf16(false, a, false, bfr,
                                                      (short)0, acc, false, false);
    }
    #pragma unroll
    for (int r = 0; r < 8; ++r) {
        long m = m0 + r + half * 8;
        int n = n0 + ln;
        float v = acc[r] + biasS[n];
        if (ACT == 2) { Cb[m * HIDD + n] = (bf16)siluf(v); }
        else          { Cf[m * HIDD + n] = v; }
    }
}

// ---------------------------------------------------------------------------
// Reset per-layer attention buffers: segment-max to enc(-inf), den/agg to 0.
// ---------------------------------------------------------------------------
__global__ __launch_bounds__(256) void attn_init_kernel(
    unsigned* __restrict__ mbuf, float* __restrict__ den, float* __restrict__ agg)
{
    long i = (long)blockIdx.x * 256 + threadIdx.x;  // N*128 threads
    if (i < (long)NODES * 4) { mbuf[i] = 0x007FFFFFu; den[i] = 0.f; }  // enc(-inf)
    agg[i] = 0.f;
}

// ---------------------------------------------------------------------------
// Fused edge kernel: per 16-edge tile compute ee = e@We with WMMA against
// LDS-resident We^T (never materialized to HBM: saves ~2.1 GB of traffic),
// consume the D fragment in-register: s = leaky(ee + xl[src] + xr[dst]),
// logit = <s, att_head>, cross-lane reduce, store logits + atomicMax(dst).
// Block = 8 waves x 16 edges = 128 edges.
// ---------------------------------------------------------------------------
__global__ __launch_bounds__(256) void edge_logits_max_kernel(
    const bf16* __restrict__ e_bf, const float* __restrict__ We,
    const float* __restrict__ att, const float* __restrict__ xl,
    const float* __restrict__ xr, const int* __restrict__ src,
    const int* __restrict__ dst, float* __restrict__ logits,
    unsigned* __restrict__ mbuf)
{
    __shared__ bf16 WeT[HIDD * HIDD];   // 32 KB, [n*128 + k]
    __shared__ float attS[HIDD];
    int tid = threadIdx.x;
    for (int i = tid; i < HIDD * HIDD; i += 256) {
        int k = i >> 7, n = i & 127;
        WeT[n * HIDD + k] = (bf16)We[i];
    }
    if (tid < HIDD) attS[tid] = att[tid];
    __syncthreads();

    int wave = tid >> 5, lane = tid & 31;
    int ln = lane & 15, half = lane >> 4;
    long tileBase = (long)blockIdx.x * 128 + wave * 16;

    // A fragments for all 4 k-steps (rows = 16 edges of this tile)
    v16bf afrag[4];
    const bf16* arow = e_bf + (tileBase + ln) * HIDD;
    #pragma unroll
    for (int k = 0; k < 4; ++k) {
        int kb = k * 32 + half * 16;
        #pragma unroll
        for (int i = 0; i < 8; ++i) { afrag[k][i] = arow[kb + i]; afrag[k][i + 8] = arow[kb + 8 + i]; }
    }
    // edge endpoints for this lane's D rows (m = r + half*8)
    int msrc[8], mdst[8];
    #pragma unroll
    for (int r = 0; r < 8; ++r) {
        long eid = tileBase + r + half * 8;
        msrc[r] = src[eid];
        mdst[r] = dst[eid];
    }

    float plog[8][4];
    #pragma unroll
    for (int r = 0; r < 8; ++r)
        #pragma unroll
        for (int hd = 0; hd < 4; ++hd) plog[r][hd] = 0.f;

    #pragma unroll
    for (int t = 0; t < 8; ++t) {                 // 8 n-tiles cover HID=128
        v8f acc = {0.f, 0.f, 0.f, 0.f, 0.f, 0.f, 0.f, 0.f};
        #pragma unroll
        for (int k = 0; k < 4; ++k) {
            v16bf bfr;
            const bf16* bcol = &WeT[(t * 16 + ln) * HIDD + k * 32 + half * 16];
            #pragma unroll
            for (int i = 0; i < 16; ++i) bfr[i] = bcol[i];
            acc = __builtin_amdgcn_wmma_f32_16x16x32_bf16(false, afrag[k], false, bfr,
                                                          (short)0, acc, false, false);
        }
        int hd = t >> 1;                           // feature f = t*16+ln -> head f/32
        int f = t * 16 + ln;
        float av = attS[f];
        #pragma unroll
        for (int r = 0; r < 8; ++r) {              // D: vgpr r -> row r (+8 for half 1), col = ln
            float s = acc[r] + xl[(long)msrc[r] * HIDD + f] + xr[(long)mdst[r] * HIDD + f];
            s = (s > 0.f) ? s : 0.2f * s;          // leaky_relu
            plog[r][hd] += s * av;
        }
    }
    // reduce over the 16 columns held by each 16-lane half
    #pragma unroll
    for (int r = 0; r < 8; ++r)
        #pragma unroll
        for (int hd = 0; hd < 4; ++hd) {
            float v = plog[r][hd];
            v += __shfl_xor(v, 1, 32);
            v += __shfl_xor(v, 2, 32);
            v += __shfl_xor(v, 4, 32);
            v += __shfl_xor(v, 8, 32);
            plog[r][hd] = v;
        }
    // each half writes its 32 (row,head) results: 2 tasks per lane
    #pragma unroll
    for (int tsel = 0; tsel < 2; ++tsel) {
        int tt = ln + tsel * 16;
        int r = tt >> 2, hd = tt & 3;
        long eid = tileBase + r + half * 8;
        float v = plog[r][hd];
        logits[eid * 4 + hd] = v;
        atomicMax(&mbuf[(long)mdst[r] * 4 + hd], fenc(v));
    }
}

// ---------------------------------------------------------------------------
// ex = exp(logit - max[dst]); den[dst] += ex
// ---------------------------------------------------------------------------
__global__ __launch_bounds__(256) void edge_exp_kernel(
    const float* __restrict__ logits, const int* __restrict__ dst,
    const unsigned* __restrict__ mbuf, float* __restrict__ exbuf,
    float* __restrict__ den)
{
    long i = (long)blockIdx.x * 256 + threadIdx.x;  // E*4 threads
    long e = i >> 2; int hd = (int)(i & 3);
    int d = dst[e];
    float ex = __expf(logits[i] - fdec(mbuf[(long)d * 4 + hd]));
    exbuf[i] = ex;
    atomicAdd(&den[(long)d * 4 + hd], ex);
}

// ---------------------------------------------------------------------------
// agg[dst] += (ex/den[dst]) * xl[src]  (the segment_sum scatter)
// ---------------------------------------------------------------------------
__global__ __launch_bounds__(256) void edge_aggr_kernel(
    const float* __restrict__ exbuf, const float* __restrict__ den,
    const float* __restrict__ xl, const int* __restrict__ src,
    const int* __restrict__ dst, float* __restrict__ agg)
{
    long i = (long)blockIdx.x * 256 + threadIdx.x;  // E*128 threads
    long e = i >> 7; int f = (int)(i & 127); int hd = f >> 5;
    int d = dst[e], s = src[e];
    float alpha = exbuf[e * 4 + hd] / den[(long)d * 4 + hd];
    atomicAdd(&agg[(long)d * HIDD + f], alpha * xl[(long)s * HIDD + f]);
}

// ---------------------------------------------------------------------------
// h' = (residual ? h : 0) + silu(LN(agg + bias)); emit f32 + bf16 copies.
// One node per 128-thread block.
// ---------------------------------------------------------------------------
__global__ __launch_bounds__(128) void node_update_kernel(
    const float* __restrict__ agg, const float* __restrict__ bias,
    const float* __restrict__ g, const float* __restrict__ b,
    float* __restrict__ h, bf16* __restrict__ hb, int residual)
{
    __shared__ float red[HIDD];
    long n = blockIdx.x; int f = threadIdx.x;
    float v = agg[n * HIDD + f] + bias[f];
    red[f] = v; __syncthreads();
    for (int s = 64; s > 0; s >>= 1) { if (f < s) red[f] += red[f + s]; __syncthreads(); }
    float mu = red[0] * (1.0f / HIDD); __syncthreads();
    float dv = v - mu;
    red[f] = dv * dv; __syncthreads();
    for (int s = 64; s > 0; s >>= 1) { if (f < s) red[f] += red[f + s]; __syncthreads(); }
    float var = red[0] * (1.0f / HIDD);
    float y = siluf(dv * rsqrtf(var + 1e-5f) * g[f] + b[f]);
    float hn = residual ? (h[n * HIDD + f] + y) : y;
    h[n * HIDD + f] = hn;
    hb[n * HIDD + f] = (bf16)hn;
}

// ---------------------------------------------------------------------------
// LN for the heads; optional bf16 / f32 outputs.
// ---------------------------------------------------------------------------
__global__ __launch_bounds__(128) void ln_head_kernel(
    const float* __restrict__ h, const float* __restrict__ g,
    const float* __restrict__ b, bf16* __restrict__ outB, float* __restrict__ outF)
{
    __shared__ float red[HIDD];
    long n = blockIdx.x; int f = threadIdx.x;
    float v = h[n * HIDD + f];
    red[f] = v; __syncthreads();
    for (int s = 64; s > 0; s >>= 1) { if (f < s) red[f] += red[f + s]; __syncthreads(); }
    float mu = red[0] * (1.0f / HIDD); __syncthreads();
    float dv = v - mu;
    red[f] = dv * dv; __syncthreads();
    for (int s = 64; s > 0; s >>= 1) { if (f < s) red[f] += red[f + s]; __syncthreads(); }
    float var = red[0] * (1.0f / HIDD);
    float y = dv * rsqrtf(var + 1e-5f) * g[f] + b[f];
    if (outB) outB[n * HIDD + f] = (bf16)y;
    if (outF) outF[n * HIDD + f] = y;
}

// ---------------------------------------------------------------------------
// color_logits = hc1_bf16 @ ch_w2(128x10) + b2   (thread per node, W in LDS)
// ---------------------------------------------------------------------------
__global__ __launch_bounds__(256) void colors_out_kernel(
    const bf16* __restrict__ hc1, const float* __restrict__ w2,
    const float* __restrict__ b2, float* __restrict__ out)
{
    __shared__ float w2s[HIDD * 10];
    __shared__ float b2s[10];
    int tid = threadIdx.x;
    for (int i = tid; i < HIDD * 10; i += 256) w2s[i] = w2[i];
    if (tid < 10) b2s[tid] = b2[tid];
    __syncthreads();
    long n = (long)blockIdx.x * 256 + tid;
    float a[10];
    #pragma unroll
    for (int j = 0; j < 10; ++j) a[j] = b2s[j];
    for (int c = 0; c < HIDD; ++c) {
        float xv = (float)hc1[n * HIDD + c];
        #pragma unroll
        for (int j = 0; j < 10; ++j) a[j] += xv * w2s[c * 10 + j];
    }
    #pragma unroll
    for (int j = 0; j < 10; ++j) out[n * 10 + j] = a[j];
}

// ---------------------------------------------------------------------------
// fg head: silu(hf @ w1(128x32) + b1) @ w2(32) + b2 (thread per node)
// ---------------------------------------------------------------------------
__global__ __launch_bounds__(256) void fg_mlp_kernel(
    const float* __restrict__ hf, const float* __restrict__ w1,
    const float* __restrict__ b1, const float* __restrict__ w2,
    const float* __restrict__ b2, float* __restrict__ out)
{
    __shared__ float w1s[HIDD * 32];   // 16 KB
    __shared__ float w2s[32];
    __shared__ float b1s[32];
    int tid = threadIdx.x;
    for (int i = tid; i < HIDD * 32; i += 256) w1s[i] = w1[i];
    if (tid < 32) { w2s[tid] = w2[tid]; b1s[tid] = b1[tid]; }
    __syncthreads();
    long n = (long)blockIdx.x * 256 + tid;
    float s[32];
    #pragma unroll
    for (int j = 0; j < 32; ++j) s[j] = b1s[j];
    for (int c = 0; c < HIDD; ++c) {
        float xv = hf[n * HIDD + c];
        #pragma unroll
        for (int j = 0; j < 32; ++j) s[j] += xv * w1s[c * 32 + j];
    }
    float acc = b2[0];
    #pragma unroll
    for (int j = 0; j < 32; ++j) acc += siluf(s[j]) * w2s[j];
    out[n] = acc;
}

__global__ __launch_bounds__(256) void copy_kernel(
    const float* __restrict__ in, float* __restrict__ out)
{
    long i = (long)blockIdx.x * 256 + threadIdx.x;
    out[i] = in[i];
}

// ---------------------------------------------------------------------------
extern "C" void kernel_launch(void* const* d_in, const int* in_sizes, int n_in,
                              void* d_out, int out_size, void* d_ws, size_t ws_size,
                              hipStream_t stream) {
    (void)in_sizes; (void)n_in; (void)out_size; (void)ws_size;
    const float* x     = (const float*)d_in[0];
    const int*   eidx  = (const int*)d_in[1];
    const float* eattr = (const float*)d_in[2];
    const float* epw   = (const float*)d_in[3];
    const float* epb   = (const float*)d_in[4];
    // layers: 5 + l*9 + {0:Wl 1:bl 2:Wr 3:br 4:We 5:att 6:bias 7:ln_g 8:ln_b}
    auto LP = [&](int l, int k) { return (const float*)d_in[5 + l * 9 + k]; };
    const float* ch_ln_g = (const float*)d_in[41];
    const float* ch_ln_b = (const float*)d_in[42];
    const float* ch_w1   = (const float*)d_in[43];
    const float* ch_b1   = (const float*)d_in[44];
    const float* ch_w2   = (const float*)d_in[45];
    const float* ch_b2   = (const float*)d_in[46];
    const float* fg_ln_g = (const float*)d_in[47];
    const float* fg_ln_b = (const float*)d_in[48];
    const float* fg_w1   = (const float*)d_in[49];
    const float* fg_b1   = (const float*)d_in[50];
    const float* fg_w2   = (const float*)d_in[51];
    const float* fg_b2   = (const float*)d_in[52];

    const int* src = eidx;
    const int* dst = eidx + EDGES;

    // workspace carve-up
    char* w = (char*)d_ws;
    bf16*     e_bf   = (bf16*)w;     w += (size_t)EDGES * HIDD * 2;
    float*    xl     = (float*)w;    w += (size_t)NODES * HIDD * 4;
    float*    xr     = (float*)w;    w += (size_t)NODES * HIDD * 4;
    float*    agg    = (float*)w;    w += (size_t)NODES * HIDD * 4;
    float*    h      = (float*)w;    w += (size_t)NODES * HIDD * 4;
    bf16*     h_bf   = (bf16*)w;     w += (size_t)NODES * HIDD * 2;
    unsigned* mbuf   = (unsigned*)w; w += (size_t)NODES * 4 * 4;
    float*    den    = (float*)w;    w += (size_t)NODES * 4 * 4;
    float*    lgts   = (float*)w;    w += (size_t)EDGES * 4 * 4;
    float*    exbuf  = (float*)w;    w += (size_t)EDGES * 4 * 4;
    bf16*     hc_bf  = (bf16*)w;     w += (size_t)NODES * HIDD * 2;
    bf16*     hc1_bf = (bf16*)w;     w += (size_t)NODES * HIDD * 2;
    float*    hf     = (float*)w;    w += (size_t)NODES * HIDD * 4;

    float* out_colors = (float*)d_out;                       // N x 10
    float* out_fg     = (float*)d_out + (size_t)NODES * 10;  // N x 1
    float* out_h      = (float*)d_out + (size_t)NODES * 11;  // N x 128

    dim3 blk256(256), blk128(128);
    const int gEdgeF = (int)((size_t)EDGES * HIDD / 256);   // 262144
    const int gNodeF = (int)((size_t)NODES * HIDD / 256);   // 32768
    const int gGemmN = NODES / 16;                          // 4096
    const int gEdgeT = EDGES / 128;                         // 4096

    edge_proj_kernel<<<gEdgeF, blk256, 0, stream>>>(eattr, epw, epb, e_bf);
    node_proj12_kernel<<<gNodeF, blk256, 0, stream>>>(x, LP(0,0), LP(0,1), LP(0,2), LP(0,3), xl, xr);

    for (int l = 0; l < 4; ++l) {
        if (l > 0) {
            gemm128_wmma_kernel<0><<<gGemmN, blk256, 0, stream>>>(h_bf, LP(l,0), LP(l,1), xl, nullptr);
            gemm128_wmma_kernel<0><<<gGemmN, blk256, 0, stream>>>(h_bf, LP(l,2), LP(l,3), xr, nullptr);
        }
        attn_init_kernel<<<gNodeF, blk256, 0, stream>>>(mbuf, den, agg);
        edge_logits_max_kernel<<<gEdgeT, blk256, 0, stream>>>(e_bf, LP(l,4), LP(l,5),
                                                              xl, xr, src, dst, lgts, mbuf);
        edge_exp_kernel<<<(int)((size_t)EDGES * 4 / 256), blk256, 0, stream>>>(lgts, dst, mbuf, exbuf, den);
        edge_aggr_kernel<<<gEdgeF, blk256, 0, stream>>>(exbuf, den, xl, src, dst, agg);
        node_update_kernel<<<NODES, blk128, 0, stream>>>(agg, LP(l,6), LP(l,7), LP(l,8),
                                                         h, h_bf, l > 0 ? 1 : 0);
    }

    // heads
    ln_head_kernel<<<NODES, blk128, 0, stream>>>(h, ch_ln_g, ch_ln_b, hc_bf, nullptr);
    gemm128_wmma_kernel<2><<<gGemmN, blk256, 0, stream>>>(hc_bf, ch_w1, ch_b1, nullptr, hc1_bf);
    colors_out_kernel<<<NODES / 256, blk256, 0, stream>>>(hc1_bf, ch_w2, ch_b2, out_colors);

    ln_head_kernel<<<NODES, blk128, 0, stream>>>(h, fg_ln_g, fg_ln_b, nullptr, hf);
    fg_mlp_kernel<<<NODES / 256, blk256, 0, stream>>>(hf, fg_w1, fg_b1, fg_w2, fg_b2, out_fg);

    copy_kernel<<<gNodeF, blk256, 0, stream>>>(h, out_h);
}
